// RealQuantLinear_53687091200599
// MI455X (gfx1250) — matrix operand order repaired and checked by reading, not compiled, not run on hardware
//
#include <hip/hip_runtime.h>

typedef __bf16 bf16;
typedef bf16 v16bf __attribute__((ext_vector_type(16)));
typedef bf16 v8bf  __attribute__((ext_vector_type(8)));
typedef bf16 v4bf  __attribute__((ext_vector_type(4)));
typedef float v8f  __attribute__((ext_vector_type(8)));

#define BM 128
#define BN 128
#define BK 32
#define K_DIM 4096
#define N_DIM 11008
#define NG 32            // n_groups = 4096/128
#define LDS_STRIDE 40    // BK + 8 pad (bf16 elems) = 80 bytes/row, 16B aligned

__global__ __launch_bounds__(256)
void qlin_wmma_bf16_kernel(const float* __restrict__ x,
                           const int*   __restrict__ w,
                           const float* __restrict__ scales,
                           const float* __restrict__ zeros,
                           const float* __restrict__ bias,
                           float* __restrict__ out)
{
    __shared__ bf16 Xs[BM * LDS_STRIDE];
    __shared__ bf16 Ws[BN * LDS_STRIDE];

    const int tid  = threadIdx.x;
    const int lane = tid & 31;
    const int wave = tid >> 5;
    const int wm   = wave >> 2;      // 0..1  (64-row band)
    const int wn   = wave & 3;       // 0..3  (32-col band)

    const int bm = blockIdx.y * BM;
    const int bn = blockIdx.x * BN;

    // cooperative tile-load mapping: 8 threads per row (4 elems each), 32 rows/pass
    const int lr = tid >> 3;         // 0..31
    const int lc = (tid & 7) * 4;    // 0,4,...,28

    // ---- prefetch registers (global -> reg, consumed next iteration) ----
    float4 xr[4];
    int4   wq[4];
    float  sc[4], zp[4];

    auto prefetch = [&](int k0) {
        const int kg = k0 >> 7;      // quant group index (GROUP_SIZE=128)
        #pragma unroll
        for (int i = 0; i < 4; ++i) {
            const int r = lr + 32 * i;
            xr[i] = *(const float4*)(x + (size_t)(bm + r) * K_DIM + k0 + lc);
            wq[i] = *(const int4*)  (w + (size_t)(bn + r) * K_DIM + k0 + lc);
            sc[i] = scales[(size_t)(bn + r) * NG + kg];
            zp[i] = zeros [(size_t)(bn + r) * NG + kg];
        }
    };

    prefetch(0);

    v8f acc[4][2];
    #pragma unroll
    for (int tm = 0; tm < 4; ++tm)
        #pragma unroll
        for (int tn = 0; tn < 2; ++tn)
            acc[tm][tn] = (v8f){0.f,0.f,0.f,0.f,0.f,0.f,0.f,0.f};

    const int lm = lane & 15;        // row (A) / col (B) within 16x16 tile
    const int kh = lane >> 4;        // K-half selector

    for (int k0 = 0; k0 < K_DIM; k0 += BK) {
        __syncthreads();             // previous compute done; safe to overwrite LDS

        // ---- reg -> LDS: x as bf16, w dequantized to bf16 ----
        #pragma unroll
        for (int i = 0; i < 4; ++i) {
            const int r = lr + 32 * i;
            v4bf xv;
            xv[0] = (bf16)xr[i].x; xv[1] = (bf16)xr[i].y;
            xv[2] = (bf16)xr[i].z; xv[3] = (bf16)xr[i].w;
            *(v4bf*)(Xs + r * LDS_STRIDE + lc) = xv;

            const float s  = sc[i];
            const float nz = -zp[i] * s;   // (q - z)*s = q*s + (-z*s)
            v4bf wv;
            wv[0] = (bf16)fmaf((float)wq[i].x, s, nz);
            wv[1] = (bf16)fmaf((float)wq[i].y, s, nz);
            wv[2] = (bf16)fmaf((float)wq[i].z, s, nz);
            wv[3] = (bf16)fmaf((float)wq[i].w, s, nz);
            *(v4bf*)(Ws + r * LDS_STRIDE + lc) = wv;
        }
        __syncthreads();

        // issue next slice's global loads before the WMMA section (pipelining)
        if (k0 + BK < K_DIM) prefetch(k0 + BK);

        // ---- LDS -> fragments (ISA 16-bit A / B layouts, wave32) ----
        v16bf afrag[4];
        #pragma unroll
        for (int tm = 0; tm < 4; ++tm) {
            const bf16* p = Xs + (wm * 64 + tm * 16 + lm) * LDS_STRIDE;
            v8bf lo = *(const v8bf*)(p + 8 * kh);        // K = 8*kh .. +7
            v8bf hi = *(const v8bf*)(p + 16 + 8 * kh);   // K = 16+8*kh .. +7
            afrag[tm] = __builtin_shufflevector(lo, hi,
                0,1,2,3,4,5,6,7,8,9,10,11,12,13,14,15);
        }
        v16bf bfrag[2];
        #pragma unroll
        for (int tn = 0; tn < 2; ++tn) {
            const bf16* p = Ws + (wn * 32 + tn * 16 + lm) * LDS_STRIDE + 8 * kh;
            v8bf lo = *(const v8bf*)(p);                 // K = 8*kh .. +7
            v8bf hi = *(const v8bf*)(p + 16);            // K = 16+8*kh .. +7
            bfrag[tn] = __builtin_shufflevector(lo, hi,
                0,1,2,3,4,5,6,7,8,9,10,11,12,13,14,15);
        }

        // ---- 8x V_WMMA_F32_16X16X32_BF16 ----
        #pragma unroll
        for (int tm = 0; tm < 4; ++tm)
            #pragma unroll
            for (int tn = 0; tn < 2; ++tn)
                acc[tm][tn] = __builtin_amdgcn_wmma_f32_16x16x32_bf16(
                    /*neg_a=*/false, afrag[tm],
                    /*neg_b=*/false, bfrag[tn],
                    /*c_mod=*/(short)0, acc[tm][tn],
                    /*reuse_a=*/false, /*reuse_b=*/false);
    }

    // ---- epilogue: + bias, store f32 ----
    #pragma unroll
    for (int tn = 0; tn < 2; ++tn) {
        const int col = bn + wn * 32 + tn * 16 + lm;
        const float bv = bias[col];
        #pragma unroll
        for (int tm = 0; tm < 4; ++tm) {
            #pragma unroll
            for (int r = 0; r < 8; ++r) {
                const int row = bm + wm * 64 + tm * 16 + r + 8 * kh;
                out[(size_t)row * N_DIM + col] = acc[tm][tn][r] + bv;
            }
        }
    }
}

extern "C" void kernel_launch(void* const* d_in, const int* in_sizes, int n_in,
                              void* d_out, int out_size, void* d_ws, size_t ws_size,
                              hipStream_t stream) {
    const float* x      = (const float*)d_in[0];
    const int*   weight = (const int*)  d_in[1];
    const float* scales = (const float*)d_in[2];
    const float* zeros  = (const float*)d_in[3];
    const float* bias   = (const float*)d_in[4];
    float* out = (float*)d_out;

    const int M = in_sizes[0] / K_DIM;          // 4096 tokens
    dim3 grid(N_DIM / BN, M / BM);              // (86, 32)
    dim3 block(256);
    qlin_wmma_bf16_kernel<<<grid, block, 0, stream>>>(x, weight, scales, zeros, bias, out);
}
